// SarcasmDetector_65429531787730
// MI455X (gfx1250) — compile-verified, matrix operand order checked by reading, not run on hardware
//
#include <hip/hip_runtime.h>
#include <hip/hip_bf16.h>
#include <cstdint>

// ---------------------------------------------------------------------------
// Types for CDNA5 WMMA (gfx1250, wave32)
// ---------------------------------------------------------------------------
typedef __bf16 bf16_t;
typedef bf16_t v16bf __attribute__((ext_vector_type(16)));
typedef float  v8f   __attribute__((ext_vector_type(8)));

union Frag { v16bf v; int4 q[2]; };

#define WMMA_BF16(A_, B_, C_) \
    __builtin_amdgcn_wmma_f32_16x16x32_bf16(false, (A_), false, (B_), (short)0, (C_), false, false)

// gfx1250 async global->LDS path (ASYNCcnt-tracked), guarded so absence of the
// builtin falls back to load/ds_store staging.
// Probe result: param 0 is `int __vector(4) __device__ *` (AS1 pointer to a
// 16-byte int vector), so cast both pointers to AS-qualified vector pointers.
#if defined(__HIP_DEVICE_COMPILE__) && defined(__gfx1250__) && \
    __has_builtin(__builtin_amdgcn_global_load_async_to_lds_b128)
#define HAS_ASYNC_LDS 1
typedef int v4i_t __attribute__((vector_size(16)));
#define GPTR(p) ((__attribute__((address_space(1))) v4i_t*)(uintptr_t)(p))
#define LPTR(p) ((__attribute__((address_space(3))) v4i_t*)(p))
#else
#define HAS_ASYNC_LDS 0
#endif

__device__ inline void wait_async_zero() {
#if HAS_ASYNC_LDS
#if __has_builtin(__builtin_amdgcn_s_wait_asynccnt)
    __builtin_amdgcn_s_wait_asynccnt(0);
#else
    asm volatile("s_wait_asynccnt 0x0" ::: "memory");
#endif
#endif
}

// Problem constants
#define BATCH 4
#define TOK   127
#define EDIM  384
#define NN    128          // nodes per batch (T+1)
#define DMODEL 128
#define HEADS_ 8
#define DH    64
#define INNER 512
#define ROWS  512          // BATCH * NN
#define LAYERS 6

// GEMM tile
#define BM 128
#define BN_T 128
#define BK 32
#define BKP 40             // padded LDS stride (40*2B = 80B, multiple of 16B)

// flags
#define F_BTRANS 1
#define F_CADD   2

__device__ inline float warp_sum(float v) {
    #pragma unroll
    for (int o = 16; o > 0; o >>= 1) v += __shfl_xor(v, o, 32);
    return v;
}
__device__ inline float warp_max(float v) {
    #pragma unroll
    for (int o = 16; o > 0; o >>= 1) v = fmaxf(v, __shfl_xor(v, o, 32));
    return v;
}

// ---------------------------------------------------------------------------
// Generic bf16 WMMA GEMM: C[M,N] = A[M,K] * B[K,N] (+bias) (+C)
// 256 threads = 8 waves; wave w computes rows [w*16, w*16+16) x 128 cols.
// Batched via grid.z with two-level strides (z1 = z % nz1, z2 = z / nz1).
// ---------------------------------------------------------------------------
__global__ __launch_bounds__(256) void gemm_bf16_kernel(
    const bf16_t* __restrict__ A, const bf16_t* __restrict__ B,
    const float* __restrict__ bias,
    float* __restrict__ Cf, bf16_t* __restrict__ Cb,
    int M, int N, int K, int lda, int ldb, int ldc,
    int nz1,
    long aS1, long aS2, long bS1, long bS2, long cS1, long cS2,
    long biasS1, long biasS2, int flags)
{
    const int z  = blockIdx.z;
    const int z1 = z % nz1;
    const int z2 = z / nz1;
    A += (long)z1 * aS1 + (long)z2 * aS2;
    B += (long)z1 * bS1 + (long)z2 * bS2;
    const long coff = (long)z1 * cS1 + (long)z2 * cS2;
    if (Cf) Cf += coff;
    if (Cb) Cb += coff;
    if (bias) bias += (long)z1 * biasS1 + (long)z2 * biasS2;

    __shared__ bf16_t As[BM * BKP];
    __shared__ bf16_t Bs[BN_T * BKP];

    const int tid  = threadIdx.x;
    const int wave = tid >> 5;
    const int lane = tid & 31;
    const int m0 = blockIdx.x * BM;
    const int n0 = blockIdx.y * BN_T;

    v8f acc[8];
    #pragma unroll
    for (int t = 0; t < 8; ++t)
        #pragma unroll
        for (int j = 0; j < 8; ++j) acc[t][j] = 0.0f;

    const int4 zero4 = {0, 0, 0, 0};
    const int half  = lane >> 4;
    const int mr    = lane & 15;
    const int kbase = half * 8;

    for (int k0 = 0; k0 < K; k0 += BK) {
        // ---- stage A tile (BM x BK) ----
        #pragma unroll
        for (int it = 0; it < 2; ++it) {
            int c  = tid + it * 256;        // 0..511 chunks of 8 elems
            int r  = c >> 2;                // tile row
            int kk = (c & 3) * 8;
            int gm = m0 + r;
#if HAS_ASYNC_LDS
            if (gm < M && (k0 + kk) < K) {
                __builtin_amdgcn_global_load_async_to_lds_b128(
                    GPTR(A + (long)gm * lda + k0 + kk),
                    LPTR(As + r * BKP + kk), 0, 0);
            } else {
                *(int4*)(As + r * BKP + kk) = zero4;
            }
#else
            int4 val = zero4;
            if (gm < M && (k0 + kk) < K)
                val = *(const int4*)(A + (long)gm * lda + k0 + kk);
            *(int4*)(As + r * BKP + kk) = val;
#endif
        }
        // ---- stage B tile as Bs[n][k] (transposed layout) ----
        if (flags & F_BTRANS) {
            // B given as N x K row-major
            #pragma unroll
            for (int it = 0; it < 2; ++it) {
                int c  = tid + it * 256;
                int r  = c >> 2;            // n index
                int kk = (c & 3) * 8;
                int gn = n0 + r;
#if HAS_ASYNC_LDS
                if (gn < N && (k0 + kk) < K) {
                    __builtin_amdgcn_global_load_async_to_lds_b128(
                        GPTR(B + (long)gn * ldb + k0 + kk),
                        LPTR(Bs + r * BKP + kk), 0, 0);
                } else {
                    *(int4*)(Bs + r * BKP + kk) = zero4;
                }
#else
                int4 val = zero4;
                if (gn < N && (k0 + kk) < K)
                    val = *(const int4*)(B + (long)gn * ldb + k0 + kk);
                *(int4*)(Bs + r * BKP + kk) = val;
#endif
            }
        } else {
            // B given as K x N row-major: read row chunks, scatter-transpose
            #pragma unroll
            for (int it = 0; it < 2; ++it) {
                int c  = tid + it * 256;    // 32 k * 16 n-chunks
                int kk = c >> 4;
                int nn = (c & 15) * 8;
                int gk = k0 + kk;
                int gn = n0 + nn;
                bf16_t tmp[8];
                if (gk < K && gn + 7 < N) {
                    int4 val = *(const int4*)(B + (long)gk * ldb + gn);
                    *(int4*)tmp = val;
                } else {
                    #pragma unroll
                    for (int j = 0; j < 8; ++j)
                        tmp[j] = (gk < K && gn + j < N)
                                   ? B[(long)gk * ldb + gn + j] : (bf16_t)0.0f;
                }
                #pragma unroll
                for (int j = 0; j < 8; ++j) Bs[(nn + j) * BKP + kk] = tmp[j];
            }
        }
        // speculative prefetch of the next K tile (global_prefetch_b8)
        if (k0 + BK < K) {
            __builtin_prefetch(A + (long)(m0 + (tid >> 1)) * lda + k0 + BK, 0, 1);
            if (flags & F_BTRANS)
                __builtin_prefetch(B + (long)(n0 + (tid >> 1)) * ldb + k0 + BK, 0, 1);
            else
                __builtin_prefetch(B + (long)(k0 + BK + (tid & 31)) * ldb + n0, 0, 1);
        }
        wait_async_zero();
        __syncthreads();

        // ---- compute: one A frag + 8 preloaded B frags, 8 back-to-back WMMAs
        const int arow = wave * 16 + mr;
        Frag af;
        af.q[0] = *(const int4*)(As + arow * BKP + kbase);
        af.q[1] = *(const int4*)(As + arow * BKP + kbase + 16);
        Frag bfg[8];
        #pragma unroll
        for (int nt = 0; nt < 8; ++nt) {
            const int brow = nt * 16 + mr;
            bfg[nt].q[0] = *(const int4*)(Bs + brow * BKP + kbase);
            bfg[nt].q[1] = *(const int4*)(Bs + brow * BKP + kbase + 16);
        }
        #pragma unroll
        for (int nt = 0; nt < 8; ++nt)
            acc[nt] = WMMA_BF16(af.v, bfg[nt].v, acc[nt]);
        __syncthreads();
    }

    // ---- epilogue ----
    const int ncol = lane & 15;
    const int mbase = wave * 16 + half * 8;
    #pragma unroll
    for (int nt = 0; nt < 8; ++nt) {
        int gn = n0 + nt * 16 + ncol;
        if (gn >= N) continue;
        float bv = bias ? bias[gn] : 0.0f;
        #pragma unroll
        for (int r = 0; r < 8; ++r) {
            int gm = m0 + mbase + r;
            if (gm >= M) continue;
            float v = acc[nt][r] + bv;
            long idx = (long)gm * ldc + gn;
            if (flags & F_CADD) v += Cf[idx];
            if (Cf) Cf[idx] = v;
            if (Cb) Cb[idx] = (bf16_t)v;
        }
    }
}

// ---------------------------------------------------------------------------
// Small-M (8 rows, padded to 16) WMMA kernel. One wave per (b,i) pair.
// Computes C[b,i][m,n] (+=) A[b,i][m, 0:128] * Bt[b,i][n, 0:128]^T (+ addRow[m])
// Fragments loaded straight from global (no LDS needed).
// Used for: scores += qWs . edges^T  and  ae = attn @ edges (via edgesT).
// ---------------------------------------------------------------------------
__global__ __launch_bounds__(256) void smallm_kernel(
    const bf16_t* __restrict__ A, long aSB, long aSI, long aSM,
    const bf16_t* __restrict__ Bt, long bSB, long bSI,
    float* __restrict__ C, long cSB, long cSI, long cSM,
    const float* __restrict__ addRow, int accumulate)
{
    const int wave = threadIdx.x >> 5;
    const int lane = threadIdx.x & 31;
    const int p = blockIdx.x * 8 + wave;     // 0..511 : (b,i)
    const int b = p >> 7;
    const int i = p & 127;

    const bf16_t* Ab = A  + (long)b * aSB + (long)i * aSI;
    const bf16_t* Bb = Bt + (long)b * bSB + (long)i * bSI;
    float*        Cp = C  + (long)b * cSB + (long)i * cSI;

    const int half  = lane >> 4;
    const int mr    = lane & 15;
    const int kbase = half * 8;
    const int4 zero4 = {0, 0, 0, 0};

    v8f acc[8];
    #pragma unroll
    for (int t = 0; t < 8; ++t)
        #pragma unroll
        for (int j = 0; j < 8; ++j) acc[t][j] = 0.0f;

    #pragma unroll
    for (int ks = 0; ks < 4; ++ks) {
        const int k0 = ks * 32;
        Frag af;
        if (mr < 8) {
            af.q[0] = *(const int4*)(Ab + (long)mr * aSM + k0 + kbase);
            af.q[1] = *(const int4*)(Ab + (long)mr * aSM + k0 + kbase + 16);
        } else {
            af.q[0] = zero4; af.q[1] = zero4;
        }
        Frag bfg[8];
        #pragma unroll
        for (int nt = 0; nt < 8; ++nt) {
            const int n = nt * 16 + mr;
            bfg[nt].q[0] = *(const int4*)(Bb + (long)n * 128 + k0 + kbase);
            bfg[nt].q[1] = *(const int4*)(Bb + (long)n * 128 + k0 + kbase + 16);
        }
        #pragma unroll
        for (int nt = 0; nt < 8; ++nt)
            acc[nt] = WMMA_BF16(af.v, bfg[nt].v, acc[nt]);
    }

    if (half == 0) {   // lanes 0..15 hold rows m=0..7 (valid head rows)
        #pragma unroll
        for (int nt = 0; nt < 8; ++nt) {
            const int n = nt * 16 + mr;
            #pragma unroll
            for (int r = 0; r < 8; ++r) {
                float v = acc[nt][r];
                long idx = (long)r * cSM + n;
                if (accumulate) v += Cp[idx];
                if (addRow)     v += addRow[((long)p << 3) + r];
                Cp[idx] = v;
            }
        }
    }
}

// ---------------------------------------------------------------------------
// Elementwise / reduction kernels (wave32)
// ---------------------------------------------------------------------------
__global__ void cvt_flat_kernel(const float* __restrict__ s, bf16_t* __restrict__ d, long n) {
    long i = (long)blockIdx.x * blockDim.x + threadIdx.x;
    long st = (long)gridDim.x * blockDim.x;
    for (; i < n; i += st) d[i] = (bf16_t)s[i];
}

__global__ void cvt_strided_kernel(const float* __restrict__ s, long lds,
                                   bf16_t* __restrict__ d, long ldd, int cols, long total) {
    long i = (long)blockIdx.x * blockDim.x + threadIdx.x;
    long st = (long)gridDim.x * blockDim.x;
    for (; i < total; i += st) {
        long r = i / cols; int c = (int)(i - r * cols);
        d[r * ldd + c] = (bf16_t)s[r * lds + c];
    }
}

// Build bf16 node-input matrix (ROWS x EDIM): row (b,0)=sent, (b,n)=tok[b,n-1]
__global__ void build_node_kernel(const float* __restrict__ sent,
                                  const float* __restrict__ tok,
                                  bf16_t* __restrict__ out) {
    long i = (long)blockIdx.x * blockDim.x + threadIdx.x;
    long st = (long)gridDim.x * blockDim.x;
    const long total = (long)ROWS * EDIM;
    for (; i < total; i += st) {
        int row = (int)(i / EDIM);
        int c   = (int)(i - (long)row * EDIM);
        int b = row >> 7, n = row & 127;
        float v = (n == 0) ? sent[(long)b * EDIM + c]
                           : tok[((long)b * TOK + (n - 1)) * EDIM + c];
        out[i] = (bf16_t)v;
    }
}

// LayerNorm over last dim 128, one wave per row, bf16 output
__global__ void ln_kernel(const float* __restrict__ x, const float* __restrict__ g,
                          const float* __restrict__ be, bf16_t* __restrict__ out, int rows) {
    int wave = threadIdx.x >> 5, lane = threadIdx.x & 31;
    int row = blockIdx.x * 8 + wave;
    if (row >= rows) return;
    const float* xr = x + (long)row * 128;
    float v[4]; float s = 0.f;
    #pragma unroll
    for (int t = 0; t < 4; ++t) { v[t] = xr[lane + t * 32]; s += v[t]; }
    float mean = warp_sum(s) * (1.0f / 128.0f);
    float q = 0.f;
    #pragma unroll
    for (int t = 0; t < 4; ++t) { float d = v[t] - mean; q += d * d; }
    float rs = rsqrtf(warp_sum(q) * (1.0f / 128.0f) + 1e-5f);
    #pragma unroll
    for (int t = 0; t < 4; ++t) {
        int c = lane + t * 32;
        out[(long)row * 128 + c] = (bf16_t)((v[t] - mean) * rs * g[c] + be[c]);
    }
}

// Edge hidden: H[b,i,j,:] = relu(LN(U[b,i]+V[b,j]+eb1; eg,ebeta)) as bf16
__global__ void edge_h_kernel(const float* __restrict__ U, const float* __restrict__ V,
                              const float* __restrict__ eb1, const float* __restrict__ eg,
                              const float* __restrict__ ebeta, bf16_t* __restrict__ H) {
    int wave = threadIdx.x >> 5, lane = threadIdx.x & 31;
    int row = blockIdx.x * 8 + wave;              // 0..65535
    int b = row >> 14, ij = row & 16383, i = ij >> 7, j = ij & 127;
    const float* u = U + ((long)b * 128 + i) * 128;
    const float* v = V + ((long)b * 128 + j) * 128;
    float val[4]; float s = 0.f;
    #pragma unroll
    for (int t = 0; t < 4; ++t) {
        int c = lane + t * 32;
        val[t] = u[c] + v[c] + eb1[c]; s += val[t];
    }
    float mean = warp_sum(s) * (1.0f / 128.0f);
    float q = 0.f;
    #pragma unroll
    for (int t = 0; t < 4; ++t) { float d = val[t] - mean; q += d * d; }
    float rs = rsqrtf(warp_sum(q) * (1.0f / 128.0f) + 1e-5f);
    #pragma unroll
    for (int t = 0; t < 4; ++t) {
        int c = lane + t * 32;
        float y = (val[t] - mean) * rs * eg[c] + ebeta[c];
        H[(long)row * 128 + c] = (bf16_t)fmaxf(0.0f, y);
    }
}

// RoPE: dst = bf16(rot(src_row) * scale); rows of width 512 (=H*Dh), ld = lds
__global__ void rope_kernel(const float* __restrict__ src, int lds,
                            bf16_t* __restrict__ dst, float scale, long total) {
    long i = (long)blockIdx.x * blockDim.x + threadIdx.x;
    long st = (long)gridDim.x * blockDim.x;
    for (; i < total; i += st) {
        int row = (int)(i >> 9);
        int col = (int)(i & 511);
        int n = row & 127;
        int d = col & 63;
        float inv = __powf(10000.0f, -(float)(d & ~1) / 64.0f);
        float ang = (float)n * inv;
        float c = __cosf(ang), snn = __sinf(ang);
        const float* sr = src + (long)row * lds;
        float x  = sr[col];
        float xp = (d & 1) ? sr[col - 1] : sr[col + 1];
        float r  = (d & 1) ? (x * c + xp * snn) : (x * c - xp * snn);
        dst[i] = (bf16_t)(r * scale);
    }
}

// qBe[b,i,h] = sum_d qs[b,i,h,d] * Be[h*64+d]
__global__ void qbe_kernel(const bf16_t* __restrict__ qs, const float* __restrict__ Be,
                           float* __restrict__ qBe) {
    int idx = blockIdx.x * blockDim.x + threadIdx.x;
    if (idx >= ROWS * HEADS_) return;
    int h = idx & 7, bi = idx >> 3;
    const bf16_t* q = qs + (long)bi * INNER + h * DH;
    const float* beh = Be + h * DH;
    float s = 0.f;
    for (int d = 0; d < DH; ++d) s += (float)q[d] * beh[d];
    qBe[idx] = s;
}

// softmax over last dim 128 (mask is all-true), writes bf16
__global__ void softmax_kernel(const float* __restrict__ sim, bf16_t* __restrict__ attn, int rows) {
    int wave = threadIdx.x >> 5, lane = threadIdx.x & 31;
    int row = blockIdx.x * 8 + wave;
    if (row >= rows) return;
    const float* sr = sim + (long)row * 128;
    float v[4]; float m = -3.0e38f;
    #pragma unroll
    for (int t = 0; t < 4; ++t) { v[t] = sr[lane + t * 32]; m = fmaxf(m, v[t]); }
    m = warp_max(m);
    float s = 0.f;
    #pragma unroll
    for (int t = 0; t < 4; ++t) { v[t] = __expf(v[t] - m); s += v[t]; }
    float inv = 1.0f / warp_sum(s);
    #pragma unroll
    for (int t = 0; t < 4; ++t)
        attn[(long)row * 128 + lane + t * 32] = (bf16_t)(v[t] * inv);
}

// zero the diagonal edge rows edges[b,i,i,:]
__global__ void zdiag_kernel(bf16_t* __restrict__ edges) {
    int idx = blockIdx.x * blockDim.x + threadIdx.x;
    if (idx >= ROWS * 128) return;
    int c = idx & 127, bi = idx >> 7;
    int b = bi >> 7, i = bi & 127;
    edges[(((long)(b * 128 + i) * 128) + i) * 128 + c] = (bf16_t)0.0f;
}

// edgesT[b,i,c,j] = edges[b,i,j,c]
__global__ void etr_kernel(const bf16_t* __restrict__ e, bf16_t* __restrict__ et, long total) {
    long i = (long)blockIdx.x * blockDim.x + threadIdx.x;
    long st = (long)gridDim.x * blockDim.x;
    for (; i < total; i += st) {
        int j = (int)(i & 127);
        int c = (int)((i >> 7) & 127);
        long bi = i >> 14;
        et[i] = e[(bi << 14) + ((long)j << 7) + c];
    }
}

// gated residual: g=sigmoid([x,res,x-res]@w); res = x*g + res*(1-g)  (in place on nodes)
__global__ void gres_kernel(const float* __restrict__ xo, float* __restrict__ nodes,
                            const float* __restrict__ w) {
    int wave = threadIdx.x >> 5, lane = threadIdx.x & 31;
    int row = blockIdx.x * 8 + wave;
    if (row >= ROWS) return;
    float a[4], nd[4]; float dot = 0.f;
    #pragma unroll
    for (int t = 0; t < 4; ++t) {
        int c = lane + t * 32;
        a[t]  = xo[(long)row * 128 + c];
        nd[t] = nodes[(long)row * 128 + c];
        dot += a[t] * w[c] + nd[t] * w[128 + c] + (a[t] - nd[t]) * w[256 + c];
    }
    dot = warp_sum(dot);
    float g = 1.0f / (1.0f + __expf(-dot));
    #pragma unroll
    for (int t = 0; t < 4; ++t) {
        int c = lane + t * 32;
        nodes[(long)row * 128 + c] = a[t] * g + nd[t] * (1.0f - g);
    }
}

__global__ void gelu_kernel(const float* __restrict__ s, bf16_t* __restrict__ d, long n) {
    long i = (long)blockIdx.x * blockDim.x + threadIdx.x;
    long st = (long)gridDim.x * blockDim.x;
    for (; i < n; i += st) {
        float x = s[i];
        d[i] = (bf16_t)(0.5f * x * (1.0f + erff(x * 0.70710678f)));
    }
}

// classifier head: one wave per batch element
__global__ void cls_kernel(const float* __restrict__ nodes, const float* __restrict__ cw1,
                           const float* __restrict__ cb1, const float* __restrict__ cg,
                           const float* __restrict__ cbeta, const float* __restrict__ cw2,
                           const float* __restrict__ cb2, float* __restrict__ out) {
    int b = blockIdx.x, lane = threadIdx.x;
    const float* sn = nodes + (long)b * NN * 128;   // row (b, 0)
    float h[4];
    #pragma unroll
    for (int t = 0; t < 4; ++t) {
        int c = lane + t * 32;
        float s = cb1[c];
        for (int k = 0; k < 128; ++k) s += sn[k] * cw1[(long)k * 128 + c];
        h[t] = s;
    }
    float s = 0.f;
    #pragma unroll
    for (int t = 0; t < 4; ++t) s += h[t];
    float mean = warp_sum(s) * (1.0f / 128.0f);
    float q = 0.f;
    #pragma unroll
    for (int t = 0; t < 4; ++t) { float d = h[t] - mean; q += d * d; }
    float rs = rsqrtf(warp_sum(q) * (1.0f / 128.0f) + 1e-5f);
    float o = 0.f;
    #pragma unroll
    for (int t = 0; t < 4; ++t) {
        int c = lane + t * 32;
        float r = fmaxf(0.0f, (h[t] - mean) * rs * cg[c] + cbeta[c]);
        o += r * cw2[c];
    }
    o = warp_sum(o) + cb2[0];
    if (lane == 0) out[b] = o;
}

// ---------------------------------------------------------------------------
// Host orchestration
// ---------------------------------------------------------------------------
static inline void gemm(hipStream_t s, const bf16_t* A, const bf16_t* B, const float* bias,
                        float* Cf, bf16_t* Cb, int M, int N, int K,
                        int lda, int ldb, int ldc, int nz, int nz1,
                        long aS1, long aS2, long bS1, long bS2, long cS1, long cS2,
                        long biasS1, long biasS2, int flags) {
    dim3 g((M + BM - 1) / BM, (N + BN_T - 1) / BN_T, nz);
    gemm_bf16_kernel<<<g, dim3(256), 0, s>>>(A, B, bias, Cf, Cb, M, N, K, lda, ldb, ldc,
                                             nz1 < 1 ? 1 : nz1, aS1, aS2, bS1, bS2,
                                             cS1, cS2, biasS1, biasS2, flags);
}

static inline int gs_blocks(long n) {
    long b = (n + 255) / 256;
    return (int)(b > 4096 ? 4096 : b);
}

extern "C" void kernel_launch(void* const* d_in, const int* in_sizes, int n_in,
                              void* d_out, int out_size, void* d_ws, size_t ws_size,
                              hipStream_t stream) {
    (void)in_sizes; (void)n_in; (void)out_size; (void)ws_size;
    const float* sent_emb = (const float*)d_in[0];
    const float* tok_emb  = (const float*)d_in[1];
    const float* node_W   = (const float*)d_in[2];
    const float* node_b   = (const float*)d_in[3];
    const float* ew1      = (const float*)d_in[4];
    const float* eb1      = (const float*)d_in[5];
    const float* eg       = (const float*)d_in[6];
    const float* ebeta    = (const float*)d_in[7];
    const float* ew2      = (const float*)d_in[8];
    const float* eb2      = (const float*)d_in[9];
    const float* ln1_g    = (const float*)d_in[10];
    const float* ln1_b    = (const float*)d_in[11];
    const float* wq       = (const float*)d_in[12];
    const float* bq       = (const float*)d_in[13];
    const float* wkv      = (const float*)d_in[14];
    const float* bkv      = (const float*)d_in[15];
    const float* we       = (const float*)d_in[16];
    const float* be       = (const float*)d_in[17];
    const float* wo       = (const float*)d_in[18];
    const float* bo       = (const float*)d_in[19];
    const float* gr1_w    = (const float*)d_in[20];
    const float* ln2_g    = (const float*)d_in[21];
    const float* ln2_b    = (const float*)d_in[22];
    const float* ffw1     = (const float*)d_in[23];
    const float* ffb1     = (const float*)d_in[24];
    const float* ffw2     = (const float*)d_in[25];
    const float* ffb2     = (const float*)d_in[26];
    const float* gr2_w    = (const float*)d_in[27];
    const float* cw1      = (const float*)d_in[28];
    const float* cb1      = (const float*)d_in[29];
    const float* cg       = (const float*)d_in[30];
    const float* cbeta    = (const float*)d_in[31];
    const float* cw2      = (const float*)d_in[32];
    const float* cb2      = (const float*)d_in[33];
    float* outp = (float*)d_out;

    // ---- workspace bump allocator (identical layout every call) ----
    char* ws = (char*)d_ws;
    size_t off = 0;
    auto alloc = [&](size_t bytes) -> char* {
        char* p = ws + off;
        off += (bytes + 255) & ~(size_t)255;
        return p;
    };
    bf16_t* nodeW_bf = (bf16_t*)alloc(sizeof(bf16_t) * EDIM * DMODEL);
    bf16_t* ew1_bf   = (bf16_t*)alloc(sizeof(bf16_t) * 2 * EDIM * 128);
    bf16_t* ew2_bf   = (bf16_t*)alloc(sizeof(bf16_t) * 128 * 128);
    bf16_t* wq_bf    = (bf16_t*)alloc(sizeof(bf16_t) * LAYERS * DMODEL * INNER);
    bf16_t* wkv_bf   = (bf16_t*)alloc(sizeof(bf16_t) * LAYERS * DMODEL * 2 * INNER);
    bf16_t* we_bf    = (bf16_t*)alloc(sizeof(bf16_t) * LAYERS * 128 * INNER);
    bf16_t* wo_bf    = (bf16_t*)alloc(sizeof(bf16_t) * LAYERS * INNER * DMODEL);
    bf16_t* ffw1_bf  = (bf16_t*)alloc(sizeof(bf16_t) * LAYERS * DMODEL * 4 * DMODEL);
    bf16_t* ffw2_bf  = (bf16_t*)alloc(sizeof(bf16_t) * LAYERS * 4 * DMODEL * DMODEL);
    bf16_t* anode_bf = (bf16_t*)alloc(sizeof(bf16_t) * ROWS * EDIM);
    float*  Ubuf     = (float*)alloc(sizeof(float) * ROWS * 128);
    float*  Vbuf     = (float*)alloc(sizeof(float) * ROWS * 128);
    bf16_t* Hrelu    = (bf16_t*)alloc(sizeof(bf16_t) * (size_t)ROWS * 128 * 128);
    bf16_t* edges    = (bf16_t*)alloc(sizeof(bf16_t) * (size_t)ROWS * 128 * 128);
    bf16_t* edgesT   = (bf16_t*)alloc(sizeof(bf16_t) * (size_t)ROWS * 128 * 128);
    float*  nodes    = (float*)alloc(sizeof(float) * ROWS * DMODEL);
    bf16_t* x_bf     = (bf16_t*)alloc(sizeof(bf16_t) * ROWS * DMODEL);
    float*  qf       = (float*)alloc(sizeof(float) * ROWS * INNER);
    float*  kvf      = (float*)alloc(sizeof(float) * ROWS * 2 * INNER);
    bf16_t* qs_bf    = (bf16_t*)alloc(sizeof(bf16_t) * ROWS * INNER);
    bf16_t* krot_bf  = (bf16_t*)alloc(sizeof(bf16_t) * ROWS * INNER);
    bf16_t* v_bf     = (bf16_t*)alloc(sizeof(bf16_t) * ROWS * INNER);
    bf16_t* qWs_bf   = (bf16_t*)alloc(sizeof(bf16_t) * ROWS * HEADS_ * 128);
    float*  qBe      = (float*)alloc(sizeof(float) * ROWS * HEADS_);
    float*  simbuf   = (float*)alloc(sizeof(float) * BATCH * HEADS_ * NN * NN);
    bf16_t* attn_bf  = (bf16_t*)alloc(sizeof(bf16_t) * BATCH * HEADS_ * NN * NN);
    float*  out_acc  = (float*)alloc(sizeof(float) * ROWS * INNER);
    float*  aebuf    = (float*)alloc(sizeof(float) * ROWS * HEADS_ * 128);
    bf16_t* ae_bf    = (bf16_t*)alloc(sizeof(bf16_t) * ROWS * HEADS_ * 128);
    bf16_t* out_bf   = (bf16_t*)alloc(sizeof(bf16_t) * ROWS * INNER);
    float*  xo       = (float*)alloc(sizeof(float) * ROWS * DMODEL);
    bf16_t* y_bf     = (bf16_t*)alloc(sizeof(bf16_t) * ROWS * DMODEL);
    float*  h1f      = (float*)alloc(sizeof(float) * ROWS * 4 * DMODEL);
    bf16_t* h1_bf    = (bf16_t*)alloc(sizeof(bf16_t) * ROWS * 4 * DMODEL);
    float*  y2       = (float*)alloc(sizeof(float) * ROWS * DMODEL);

    // ---- weight conversion to bf16 ----
    auto cvt = [&](const float* s, bf16_t* d, long n) {
        cvt_flat_kernel<<<gs_blocks(n), 256, 0, stream>>>(s, d, n);
    };
    cvt(node_W, nodeW_bf, (long)EDIM * DMODEL);
    cvt(ew1,    ew1_bf,   (long)2 * EDIM * 128);
    cvt(ew2,    ew2_bf,   128L * 128);
    cvt(wq,     wq_bf,    (long)LAYERS * DMODEL * INNER);
    cvt(wkv,    wkv_bf,   (long)LAYERS * DMODEL * 2 * INNER);
    cvt(we,     we_bf,    (long)LAYERS * 128 * INNER);
    cvt(wo,     wo_bf,    (long)LAYERS * INNER * DMODEL);
    cvt(ffw1,   ffw1_bf,  (long)LAYERS * DMODEL * 4 * DMODEL);
    cvt(ffw2,   ffw2_bf,  (long)LAYERS * 4 * DMODEL * DMODEL);

    // ---- node inputs and initial node embedding ----
    build_node_kernel<<<gs_blocks((long)ROWS * EDIM), 256, 0, stream>>>(sent_emb, tok_emb, anode_bf);
    // nodes = anode @ node_W + node_b
    gemm(stream, anode_bf, nodeW_bf, node_b, nodes, nullptr, ROWS, DMODEL, EDIM,
         EDIM, DMODEL, DMODEL, 1, 1, 0,0,0,0,0,0, 0,0, 0);
    // U = anode @ ew1_top ; V = anode @ ew1_bot
    gemm(stream, anode_bf, ew1_bf, nullptr, Ubuf, nullptr, ROWS, 128, EDIM,
         EDIM, 128, 128, 1, 1, 0,0,0,0,0,0, 0,0, 0);
    gemm(stream, anode_bf, ew1_bf + (long)EDIM * 128, nullptr, Vbuf, nullptr, ROWS, 128, EDIM,
         EDIM, 128, 128, 1, 1, 0,0,0,0,0,0, 0,0, 0);
    // H = relu(LN(U_i + V_j + eb1))
    edge_h_kernel<<<8192, 256, 0, stream>>>(Ubuf, Vbuf, eb1, eg, ebeta, Hrelu);
    // edges = H @ ew2 + eb2  (bf16 out)
    gemm(stream, Hrelu, ew2_bf, eb2, nullptr, edges, ROWS * 128, 128, 128,
         128, 128, 128, 1, 1, 0,0,0,0,0,0, 0,0, 0);
    zdiag_kernel<<<(ROWS * 128 + 255) / 256, 256, 0, stream>>>(edges);
    etr_kernel<<<4096, 256, 0, stream>>>(edges, edgesT, (long)ROWS * 128 * 128);

    const float scale = 0.125f;   // Dh^-0.5

    for (int l = 0; l < LAYERS; ++l) {
        const bf16_t* wq_l   = wq_bf   + (long)l * DMODEL * INNER;
        const bf16_t* wkv_l  = wkv_bf  + (long)l * DMODEL * 2 * INNER;
        const bf16_t* we_l   = we_bf   + (long)l * 128 * INNER;
        const bf16_t* wo_l   = wo_bf   + (long)l * INNER * DMODEL;
        const bf16_t* ffw1_l = ffw1_bf + (long)l * DMODEL * 4 * DMODEL;
        const bf16_t* ffw2_l = ffw2_bf + (long)l * 4 * DMODEL * DMODEL;

        // x = LN(nodes)
        ln_kernel<<<64, 256, 0, stream>>>(nodes, ln1_g + l * 128, ln1_b + l * 128, x_bf, ROWS);
        // q = x@Wq+bq ; kv = x@Wkv+bkv
        gemm(stream, x_bf, wq_l, bq + (long)l * INNER, qf, nullptr, ROWS, INNER, DMODEL,
             DMODEL, INNER, INNER, 1, 1, 0,0,0,0,0,0, 0,0, 0);
        gemm(stream, x_bf, wkv_l, bkv + (long)l * 2 * INNER, kvf, nullptr, ROWS, 2 * INNER, DMODEL,
             DMODEL, 2 * INNER, 2 * INNER, 1, 1, 0,0,0,0,0,0, 0,0, 0);
        // rotary: qs = rot(q)*scale, krot = rot(k), v = bf16(v)
        rope_kernel<<<gs_blocks((long)ROWS * INNER), 256, 0, stream>>>(qf, INNER, qs_bf, scale, (long)ROWS * INNER);
        rope_kernel<<<gs_blocks((long)ROWS * INNER), 256, 0, stream>>>(kvf, 2 * INNER, krot_bf, 1.0f, (long)ROWS * INNER);
        cvt_strided_kernel<<<gs_blocks((long)ROWS * INNER), 256, 0, stream>>>(
            kvf + INNER, 2 * INNER, v_bf, INNER, INNER, (long)ROWS * INNER);
        // qWs[b,i,h,:] = qs[b,i,h,:] @ We_h^T  (bf16 out), batched over h (K=64)
        gemm(stream, qs_bf, we_l, nullptr, nullptr, qWs_bf, ROWS, 128, DH,
             INNER, INNER, HEADS_ * 128, HEADS_, HEADS_,
             /*aS1*/DH, 0, /*bS1*/DH, 0, /*cS1*/128, 0, 0, 0, F_BTRANS);
        // qBe[b,i,h] = qs . Be_h
        qbe_kernel<<<(ROWS * HEADS_ + 255) / 256, 256, 0, stream>>>(qs_bf, be + (long)l * INNER, qBe);
        // sim1 = qs @ k^T per (b,h)
        gemm(stream, qs_bf, krot_bf, nullptr, simbuf, nullptr, NN, NN, DH,
             INNER, INNER, NN, BATCH * HEADS_, HEADS_,
             /*aS1*/DH, /*aS2*/(long)NN * INNER,
             /*bS1*/DH, /*bS2*/(long)NN * INNER,
             /*cS1*/(long)NN * NN, /*cS2*/(long)HEADS_ * NN * NN, 0, 0, F_BTRANS);
        // sim += qWs . edges^T + qBe
        smallm_kernel<<<64, 256, 0, stream>>>(
            qWs_bf, (long)NN * HEADS_ * 128, (long)HEADS_ * 128, 128,
            edges, (long)NN * NN * 128, (long)NN * 128,
            simbuf, (long)HEADS_ * NN * NN, NN, (long)NN * NN,
            qBe, 1);
        // attn = softmax(sim)
        softmax_kernel<<<512, 256, 0, stream>>>(simbuf, attn_bf, BATCH * HEADS_ * NN);
        // out1 = attn @ v per (b,h)  ->  out_acc (b,i,h*64+d)
        gemm(stream, attn_bf, v_bf, nullptr, out_acc, nullptr, NN, DH, NN,
             NN, INNER, INNER, BATCH * HEADS_, HEADS_,
             /*aS1*/(long)NN * NN, /*aS2*/(long)HEADS_ * NN * NN,
             /*bS1*/DH, /*bS2*/(long)NN * INNER,
             /*cS1*/DH, /*cS2*/(long)NN * INNER, 0, 0, 0);
        // ae[b,i,h,:] = attn[b,h,i,:] @ edges[b,i]   (via edgesT)
        smallm_kernel<<<64, 256, 0, stream>>>(
            attn_bf, (long)HEADS_ * NN * NN, NN, (long)NN * NN,
            edgesT, (long)NN * NN * 128, (long)NN * 128,
            aebuf, (long)NN * HEADS_ * 128, (long)HEADS_ * 128, 128,
            nullptr, 0);
        cvt(aebuf, ae_bf, (long)ROWS * HEADS_ * 128);
        // out_acc += ae @ We_h + Be_h  (batched over h)
        gemm(stream, ae_bf, we_l, be + (long)l * INNER, out_acc, nullptr, ROWS, DH, 128,
             HEADS_ * 128, INNER, INNER, HEADS_, HEADS_,
             /*aS1*/128, 0, /*bS1*/DH, 0, /*cS1*/DH, 0, /*biasS1*/DH, 0, F_CADD);
        // xo = out_acc @ Wo + bo
        cvt(out_acc, out_bf, (long)ROWS * INNER);
        gemm(stream, out_bf, wo_l, bo + (long)l * DMODEL, xo, nullptr, ROWS, DMODEL, INNER,
             INNER, DMODEL, DMODEL, 1, 1, 0,0,0,0,0,0, 0,0, 0);
        // nodes = gated_res(xo, nodes, gr1_w)
        gres_kernel<<<64, 256, 0, stream>>>(xo, nodes, gr1_w + (long)l * 3 * DMODEL);
        // FFN
        ln_kernel<<<64, 256, 0, stream>>>(nodes, ln2_g + l * 128, ln2_b + l * 128, y_bf, ROWS);
        gemm(stream, y_bf, ffw1_l, ffb1 + (long)l * 4 * DMODEL, h1f, nullptr, ROWS, 4 * DMODEL, DMODEL,
             DMODEL, 4 * DMODEL, 4 * DMODEL, 1, 1, 0,0,0,0,0,0, 0,0, 0);
        gelu_kernel<<<gs_blocks((long)ROWS * 4 * DMODEL), 256, 0, stream>>>(h1f, h1_bf, (long)ROWS * 4 * DMODEL);
        gemm(stream, h1_bf, ffw2_l, ffb2 + (long)l * DMODEL, y2, nullptr, ROWS, DMODEL, 4 * DMODEL,
             4 * DMODEL, DMODEL, DMODEL, 1, 1, 0,0,0,0,0,0, 0,0, 0);
        gres_kernel<<<64, 256, 0, stream>>>(y2, nodes, gr2_w + (long)l * 3 * DMODEL);
    }

    // classifier head -> 4 outputs
    cls_kernel<<<BATCH, 32, 0, stream>>>(nodes, cw1, cb1, cg, cbeta, cw2, cb2, outp);
}
